// ImageMoE_44873818308995
// MI455X (gfx1250) — matrix operand order repaired and checked by reading, compiled, not run on hardware
//
#include <hip/hip_runtime.h>
#include <hip/hip_bf16.h>
#include <math.h>

typedef __attribute__((ext_vector_type(16))) _Float16 v16h;
typedef __attribute__((ext_vector_type(8)))  _Float16 v8h;
typedef __attribute__((ext_vector_type(4)))  _Float16 v4h;
typedef __attribute__((ext_vector_type(8)))  float    v8f;
typedef __attribute__((ext_vector_type(4)))  unsigned int v4u;
typedef __attribute__((ext_vector_type(8)))  int v8i;
typedef __attribute__((ext_vector_type(4)))  int v4i;

#define Bsz 64
#define NTOK 256
#define DIM 256
#define PD 196
#define KP0 224          // PD padded to multiple of 32
#define OUTD 1024
#define HID 4096
#define NEXP 16
#define T_TOT (Bsz * NTOK)   // 16384

#if defined(__has_builtin)
#if __has_builtin(__builtin_amdgcn_tensor_load_to_lds) && \
    __has_builtin(__builtin_amdgcn_s_wait_tensorcnt)
#define USE_TDM 1
#endif
#endif

#ifdef USE_TDM
// Issue a TDM load of a 64x32 f16 tile (row-major, stride lda elements) into
// LDS at byte offset lds_off.  tensor_dim1 = valid rows => TDM zero-fills the
// M edge.  D# layout per CDNA5 ISA (group0 128b, group1 256b).
// This toolchain exposes the 6-arg builtin:
//   (uint32x4 g0, int32x8 g1, int32x4 g2, int32x4 g3, int32x8 pad, i32 cpol)
__device__ inline void tdm_load_tile_a(unsigned lds_off, const _Float16* gaddr,
                                       int lda, int rows)
{
    unsigned long long ga = (unsigned long long)(uintptr_t)gaddr;
    v4u g0 = { 1u,                                            // count=1 (valid, user mode)
               lds_off,                                        // lds_addr
               (unsigned)ga,                                   // global_addr[31:0]
               ((unsigned)(ga >> 32) & 0x1FFFFFFu) | (2u << 30) }; // ga[56:32] | type=2
    v8i g1 = { (int)(1u << 16),                                // data_size=1 (2 bytes)
               (int)(32u << 16),                               // tensor_dim0 = 32
               (int)((unsigned)rows << 16),                    // tensor_dim1 = valid rows
               (int)(32u << 16),                               // tile_dim0 = 32
               64,                                             // tile_dim1 = 64
               lda,                                            // tensor_dim0_stride
               0, 0 };                                         // tensor_dim1_stride = 0
    v4i z4 = { 0, 0, 0, 0 };
    v8i z8 = { 0, 0, 0, 0, 0, 0, 0, 0 };
    __builtin_amdgcn_tensor_load_to_lds(g0, g1, z4, z4, z8, 0);
}
#endif

// ---------------------------------------------------------------------------
// Generic f16 WMMA GEMM:  C(MxN) = epilogue(A(MxK) @ B(KxN))
//  - 64x64 tile / block (256 thr = 8 waves; wave = 16 rows x 32 cols =>
//    2x v_wmma_f32_16x16x32_f16 per K-step)
//  - A tile double-buffered via TDM (tensor_load_to_lds + s_wait_tensorcnt)
//  - B tile stored transposed in LDS so both fragments load as ds_load_b128
//  - requires N % 64 == 0 and K % 32 == 0 (true for every call site)
// ---------------------------------------------------------------------------
__global__ __launch_bounds__(256)
void k_gemm(const _Float16* __restrict__ A, int lda,
            const _Float16* __restrict__ Bm, int ldb,
            const float* __restrict__ bias, int relu,
            float* __restrict__ C, _Float16* __restrict__ C16, int ldc,
            const float* __restrict__ Radd,
            int Mconst, const int* __restrict__ Mptr,
            int N, int K,
            const int* __restrict__ scat_idx,
            const float* __restrict__ scat_gate, int gate_e,
            float* __restrict__ scat_out)
{
    __shared__ _Float16 As2[2][64 * 32];
    __shared__ _Float16 BsT[64 * 32];       // transposed: BsT[col][k]

    const int M = Mptr ? *Mptr : Mconst;
    const int m0 = blockIdx.x * 64;
    if (m0 >= M) return;                    // uniform per block
    const int n0 = blockIdx.y * 64;

    const int t     = threadIdx.x;
    const int w     = t >> 5;
    const int lane  = t & 31;
    const int half_ = lane >> 4;
    const int mr    = lane & 15;
    const int mbase = (w & 3) << 4;         // 0,16,32,48
    const int cbase = (w >> 2) << 5;        // 0,32

    const int rows   = (M - m0) < 64 ? (M - m0) : 64;
    const int nsteps = K >> 5;

#ifdef USE_TDM
    if (w == 0)                              // one TDM issue per block (wave 0)
        tdm_load_tile_a((unsigned)(uintptr_t)(void*)&As2[0][0],
                        A + (size_t)m0 * lda, lda, rows);
#endif

    v8f c0 = {}; v8f c1 = {};

    for (int i = 0; i < nsteps; ++i) {
        const int k0  = i << 5;
        const int cur = i & 1;

#ifdef USE_TDM
        if (w == 0) {
            if (i + 1 < nsteps) {
                tdm_load_tile_a((unsigned)(uintptr_t)(void*)&As2[cur ^ 1][0],
                                A + (size_t)m0 * lda + (k0 + 32), lda, rows);
                __builtin_amdgcn_s_wait_tensorcnt(1);   // current tile done
            } else {
                __builtin_amdgcn_s_wait_tensorcnt(0);
            }
        }
#else
        {   // manual A tile: 16B per thread, row-granular predication
            int ra = t >> 2, ca = (t & 3) << 3;
            v8h av = {};
            if (m0 + ra < M)
                av = *(const v8h*)(A + (size_t)(m0 + ra) * lda + k0 + ca);
            *(v8h*)(&As2[cur][ra * 32 + ca]) = av;
        }
#endif
        {   // B tile: global_load_b128, scatter-transpose into LDS
            int rb = t >> 3, cb0 = (t & 7) << 3;
            v8h bv = *(const v8h*)(Bm + (size_t)(k0 + rb) * ldb + n0 + cb0);
            #pragma unroll
            for (int j = 0; j < 8; ++j)
                BsT[(cb0 + j) * 32 + rb] = bv[j];
        }

        if (i + 1 < nsteps && t < 32)        // gfx1250 global_prefetch_b8
            __builtin_prefetch(Bm + (size_t)(k0 + 32 + t) * ldb + n0, 0, 0);

        __syncthreads();                     // tiles visible (TDM waited above)

        // A fragment: lane = row mbase+mr; K = 16*(e/8) + 8*half + e%8
        const _Float16* ap = &As2[cur][(mbase + mr) * 32 + (half_ << 3)];
        v8h alo = *(const v8h*)ap;           // K = 8*half .. +7
        v8h ahi = *(const v8h*)(ap + 16);    // K = 16+8*half .. +7
        v16h a = __builtin_shufflevector(alo, ahi,
                 0,1,2,3,4,5,6,7,8,9,10,11,12,13,14,15);

        // B fragments: lane = col cbase+mr(+16); K = 16*(e/8) + 8*half + e%8
        const _Float16* bp0 = &BsT[(cbase + mr) * 32 + (half_ << 3)];
        v8h b0l = *(const v8h*)bp0, b0h = *(const v8h*)(bp0 + 16);
        v16h b0 = __builtin_shufflevector(b0l, b0h,
                  0,1,2,3,4,5,6,7,8,9,10,11,12,13,14,15);
        const _Float16* bp1 = bp0 + 16 * 32;
        v8h b1l = *(const v8h*)bp1, b1h = *(const v8h*)(bp1 + 16);
        v16h b1 = __builtin_shufflevector(b1l, b1h,
                  0,1,2,3,4,5,6,7,8,9,10,11,12,13,14,15);

        c0 = __builtin_amdgcn_wmma_f32_16x16x32_f16(false, a, false, b0, (short)0, c0, false, false);
        c1 = __builtin_amdgcn_wmma_f32_16x16x32_f16(false, a, false, b1, (short)0, c1, false, false);
        __syncthreads();                     // done reading tiles
    }

    // epilogue: C VGPR r -> row r + 8*half, col = lane%16
    #pragma unroll
    for (int r = 0; r < 8; ++r) {
        int m = m0 + mbase + (half_ << 3) + r;
        if (m >= M) continue;
        #pragma unroll
        for (int s = 0; s < 2; ++s) {
            int n = n0 + cbase + (s << 4) + mr;
            float v = s ? c1[r] : c0[r];
            if (bias) v += bias[n];
            if (relu) v = v > 0.f ? v : 0.f;
            if (Radd) v += Radd[(size_t)m * ldc + n];
            if (scat_out) {
                int row = scat_idx[m];
                float g = scat_gate[(size_t)row * NEXP + gate_e];
                scat_out[(size_t)row * ldc + n] += g * v;
            } else {
                if (C)   C  [(size_t)m * ldc + n] = v;
                if (C16) C16[(size_t)m * ldc + n] = (_Float16)v;
            }
        }
    }
}

// ---------------------------------------------------------------------------
// helper kernels
// ---------------------------------------------------------------------------
__global__ void k_f32_to_f16(const float* __restrict__ X, _Float16* __restrict__ Y, int n4) {
    int i = blockIdx.x * 256 + threadIdx.x;          // n4 = n/4 (always exact here)
    if (i < n4) {
        float4 f = *(const float4*)(X + 4 * (size_t)i);
        v4h h = { (_Float16)f.x, (_Float16)f.y, (_Float16)f.z, (_Float16)f.w };
        *(v4h*)(Y + 4 * (size_t)i) = h;
    }
}

__global__ __launch_bounds__(224)
void k_patchify(const float* __restrict__ x, _Float16* __restrict__ xp) {
    int bn = blockIdx.x;                 // b*256 + n
    int b = bn >> 8, n = bn & 255;
    int p = threadIdx.x;                 // 0..223 (196 real + pad)
    float v = 0.f;
    if (p < PD) {
        int r = p / 14, c = p % 14;
        int py = (n >> 4) * 14 + r, px = (n & 15) * 14 + c;
        v = x[(size_t)b * 224 * 224 + py * 224 + px];
    }
    xp[(size_t)bn * KP0 + p] = (_Float16)v;
}

__global__ void k_pack_wpe(const float* __restrict__ W, _Float16* __restrict__ Y) {
    int k = blockIdx.x, d = threadIdx.x; // 224 x 256
    Y[k * DIM + d] = (_Float16)(k < PD ? W[k * DIM + d] : 0.f);
}

// (H,D,HS) -> (D, H*HS) f16
__global__ void k_pack_qkv(const float* __restrict__ W, _Float16* __restrict__ Y) {
    int d = blockIdx.x, t = threadIdx.x; // t = h*32+e
    Y[d * DIM + t] = (_Float16)W[(size_t)(t >> 5) * DIM * 32 + d * 32 + (t & 31)];
}

__global__ void k_add_pe(float* __restrict__ h) {
    int bn = blockIdx.x;  int n = bn & 255;  int d = threadIdx.x;
    int i2 = d & ~1;
    float div = __expf(-(float)i2 * (9.2103403719761836f / 256.f));
    float ang = (float)n * div;
    float pe = (d & 1) ? __cosf(ang) : __sinf(ang);
    h[(size_t)bn * DIM + d] += pe;
}

__global__ __launch_bounds__(256)
void k_layernorm(const float* __restrict__ X, _Float16* __restrict__ Y,
                 const float* __restrict__ g, const float* __restrict__ b, int L)
{
    __shared__ float red[256];
    int row = blockIdx.x;
    const float* x = X + (size_t)row * L;
    float s = 0.f;
    for (int i = threadIdx.x; i < L; i += 256) s += x[i];
    red[threadIdx.x] = s; __syncthreads();
    for (int o = 128; o > 0; o >>= 1) { if (threadIdx.x < o) red[threadIdx.x] += red[threadIdx.x + o]; __syncthreads(); }
    float mean = red[0] / L; __syncthreads();
    float v = 0.f;
    for (int i = threadIdx.x; i < L; i += 256) { float d = x[i] - mean; v += d * d; }
    red[threadIdx.x] = v; __syncthreads();
    for (int o = 128; o > 0; o >>= 1) { if (threadIdx.x < o) red[threadIdx.x] += red[threadIdx.x + o]; __syncthreads(); }
    float inv = rsqrtf(red[0] / L + 1e-5f);
    for (int i = threadIdx.x; i < L; i += 256)
        Y[(size_t)row * L + i] = (_Float16)((x[i] - mean) * inv * g[i] + b[i]);
}

// per (b,h) causal attention; K/V tiles in LDS, one query row per thread
__global__ __launch_bounds__(256)
void k_attention(const float* __restrict__ qb, const float* __restrict__ kb,
                 const float* __restrict__ vb, _Float16* __restrict__ sa)
{
    __shared__ float Ks[NTOK * 32];
    __shared__ float Vs[NTOK * 32];
    int b = blockIdx.x >> 3, hh = blockIdx.x & 7;
    int t = threadIdx.x;
    size_t base = ((size_t)b * NTOK) * DIM + hh * 32;
    #pragma unroll
    for (int e = 0; e < 32; e += 4) {
        *(float4*)&Ks[t * 32 + e] = *(const float4*)&kb[base + (size_t)t * DIM + e];
        *(float4*)&Vs[t * 32 + e] = *(const float4*)&vb[base + (size_t)t * DIM + e];
    }
    __syncthreads();
    float q[32];
    #pragma unroll
    for (int e = 0; e < 32; e += 4) {
        float4 f = *(const float4*)&qb[base + (size_t)t * DIM + e];
        q[e] = f.x; q[e + 1] = f.y; q[e + 2] = f.z; q[e + 3] = f.w;
    }
    const float scl = 0.17677669529663687f;   // 1/sqrt(32)
    float mx = -1e30f;
    for (int j = 0; j <= t; ++j) {
        float s = 0.f;
        #pragma unroll
        for (int e = 0; e < 32; ++e) s += q[e] * Ks[j * 32 + e];
        mx = fmaxf(mx, s * scl);
    }
    float sum = 0.f, o[32];
    #pragma unroll
    for (int e = 0; e < 32; ++e) o[e] = 0.f;
    for (int j = 0; j <= t; ++j) {
        float s = 0.f;
        #pragma unroll
        for (int e = 0; e < 32; ++e) s += q[e] * Ks[j * 32 + e];
        float p = __expf(s * scl - mx); sum += p;
        #pragma unroll
        for (int e = 0; e < 32; ++e) o[e] += p * Vs[j * 32 + e];
    }
    float inv = 1.f / sum;
    #pragma unroll
    for (int e = 0; e < 32; ++e)
        sa[base + (size_t)t * DIM + e] = (_Float16)(o[e] * inv);
}

__device__ inline unsigned pcg(unsigned x) {
    x = x * 747796405u + 2891336453u;
    x = ((x >> ((x >> 28) + 4u)) ^ x) * 277803737u;
    return (x >> 22) ^ x;
}
__device__ inline float hash_normal(unsigned a, unsigned b) {
    unsigned u1 = pcg(a * 0x9E3779B9u + b + 1u);
    unsigned u2 = pcg(u1 + 0x85EBCA6Bu + b * 0xC2B2AE35u);
    float f1 = ((float)(u1 >> 8) + 1.f) * (1.f / 16777217.f);
    float f2 = (float)u2 * (1.f / 4294967296.f);
    return sqrtf(-2.f * logf(f1)) * __cosf(6.283185307f * f2);
}

// noisy top-2 router + sparse softmax gate; one token per block
__global__ __launch_bounds__(256)
void k_router(const _Float16* __restrict__ X,
              const float* __restrict__ Wr, const float* __restrict__ br,
              const float* __restrict__ Wn, const float* __restrict__ bn,
              float* __restrict__ gate, unsigned seed)
{
    __shared__ float rl[256], rn_[256];
    __shared__ float noisy[NEXP];
    int tok = blockIdx.x;
    int e = threadIdx.x >> 4, i = threadIdx.x & 15;
    const _Float16* x = X + (size_t)tok * OUTD;
    float sl = 0.f, sn = 0.f;
    for (int j = i * 64; j < i * 64 + 64; ++j) {
        float xv = (float)x[j];
        sl += xv * Wr[j * NEXP + e];
        sn += xv * Wn[j * NEXP + e];
    }
    rl[threadIdx.x] = sl; rn_[threadIdx.x] = sn; __syncthreads();
    for (int o = 8; o > 0; o >>= 1) {
        if (i < o) { rl[threadIdx.x] += rl[threadIdx.x + o]; rn_[threadIdx.x] += rn_[threadIdx.x + o]; }
        __syncthreads();
    }
    if (i == 0) {
        float logit = rl[threadIdx.x] + br[e];
        float npre  = rn_[threadIdx.x] + bn[e];
        float sp = (npre > 20.f) ? npre : log1pf(__expf(npre));   // softplus
        noisy[e] = logit + hash_normal(seed + (unsigned)tok, (unsigned)e) * sp;
    }
    __syncthreads();
    if (threadIdx.x == 0) {
        int i1 = -1, i2 = -1; float v1 = -1e30f, v2 = -1e30f;
        for (int q = 0; q < NEXP; ++q) {
            float v = noisy[q];
            if (v > v1) { v2 = v1; i2 = i1; v1 = v; i1 = q; }
            else if (v > v2) { v2 = v; i2 = q; }
        }
        float e1 = 1.f, e2 = __expf(v2 - v1), den = e1 + e2;
        for (int q = 0; q < NEXP; ++q) gate[(size_t)tok * NEXP + q] = 0.f;
        gate[(size_t)tok * NEXP + i1] = e1 / den;
        gate[(size_t)tok * NEXP + i2] = e2 / den;
    }
}

__global__ void k_build_lists(const float* __restrict__ gate,
                              int* __restrict__ counts, int* __restrict__ idx)
{
    int tok = blockIdx.x * 256 + threadIdx.x;
    if (tok >= T_TOT) return;
    for (int e = 0; e < NEXP; ++e)
        if (gate[(size_t)tok * NEXP + e] > 0.f) {
            int p = atomicAdd(&counts[e], 1);
            idx[e * T_TOT + p] = tok;
        }
}

__global__ __launch_bounds__(128)
void k_gather(const _Float16* __restrict__ X, const int* __restrict__ idx,
              const int* __restrict__ count, _Float16* __restrict__ A)
{
    int m = blockIdx.x;
    if (m >= *count) return;
    int tok = idx[m];
    int j = threadIdx.x * 8;                         // 128 threads * 8 halfs = 1024
    *(v8h*)(A + (size_t)m * OUTD + j) = *(const v8h*)(X + (size_t)tok * OUTD + j);
}

__global__ void k_mean(const float* __restrict__ second, float* __restrict__ iv) {
    int b = blockIdx.x;
    int d = blockIdx.y * 256 + threadIdx.x;
    float s = 0.f;
    for (int n = 0; n < NTOK; ++n) s += second[((size_t)b * NTOK + n) * OUTD + d];
    iv[(size_t)b * OUTD + d] = s * (1.f / NTOK);
}

__global__ __launch_bounds__(256)
void k_cls(const float* __restrict__ iv, const float* __restrict__ Wc,
           const float* __restrict__ bc, float* __restrict__ cls)
{
    __shared__ float red[256];
    int b = blockIdx.x;
    float s = 0.f;
    for (int d = threadIdx.x; d < OUTD; d += 256) s += iv[(size_t)b * OUTD + d] * Wc[d];
    red[threadIdx.x] = s; __syncthreads();
    for (int o = 128; o > 0; o >>= 1) { if (threadIdx.x < o) red[threadIdx.x] += red[threadIdx.x + o]; __syncthreads(); }
    if (threadIdx.x == 0) cls[b] = red[0] + bc[0];
}

// ---------------------------------------------------------------------------
extern "C" void kernel_launch(void* const* d_in, const int* in_sizes, int n_in,
                              void* d_out, int out_size, void* d_ws, size_t ws_size,
                              hipStream_t stream)
{
    (void)in_sizes; (void)n_in; (void)out_size; (void)ws_size;

    const float* x   = (const float*)d_in[0];
    const float* Wpe = (const float*)d_in[1];
    const float* bpe = (const float*)d_in[2];
    const float* g1  = (const float*)d_in[3];
    const float* be1 = (const float*)d_in[4];
    const float* Wq  = (const float*)d_in[5];
    const float* Wk  = (const float*)d_in[6];
    const float* Wv  = (const float*)d_in[7];
    const float* Wo  = (const float*)d_in[8];
    const float* bo  = (const float*)d_in[9];
    const float* Wp  = (const float*)d_in[10];
    const float* bp  = (const float*)d_in[11];
    const float* g2  = (const float*)d_in[12];
    const float* be2 = (const float*)d_in[13];
    const float* g3  = (const float*)d_in[14];
    const float* be3 = (const float*)d_in[15];
    const float* Wc  = (const float*)d_in[16];
    const float* bc  = (const float*)d_in[17];
    const float* moe_p[2][8];
    for (int L = 0; L < 2; ++L)
        for (int j = 0; j < 8; ++j) moe_p[L][j] = (const float*)d_in[18 + L * 8 + j];

    float* out_first  = (float*)d_out;
    float* out_second = out_first + (size_t)T_TOT * OUTD;
    float* out_iv     = out_second + (size_t)T_TOT * OUTD;
    float* out_cls    = out_iv + (size_t)Bsz * OUTD;

    // workspace carve-up
    char* ws = (char*)d_ws; size_t off = 0;
    auto alloc = [&](size_t bytes) -> void* {
        void* p = ws + off; off = (off + bytes + 255) & ~(size_t)255; return p;
    };
    _Float16* xp16   = (_Float16*)alloc((size_t)T_TOT * KP0 * 2);
    _Float16* wpe16  = (_Float16*)alloc((size_t)KP0 * DIM * 2);
    float*    h      = (float*)   alloc((size_t)T_TOT * DIM * 4);
    _Float16* hn16   = (_Float16*)alloc((size_t)T_TOT * DIM * 2);
    _Float16* wq16   = (_Float16*)alloc((size_t)DIM * DIM * 2);
    _Float16* wk16   = (_Float16*)alloc((size_t)DIM * DIM * 2);
    _Float16* wv16   = (_Float16*)alloc((size_t)DIM * DIM * 2);
    float*    qb     = (float*)   alloc((size_t)T_TOT * DIM * 4);
    float*    kb     = (float*)   alloc((size_t)T_TOT * DIM * 4);
    float*    vb     = (float*)   alloc((size_t)T_TOT * DIM * 4);
    _Float16* sa16   = (_Float16*)alloc((size_t)T_TOT * DIM * 2);
    _Float16* wo16   = (_Float16*)alloc((size_t)DIM * DIM * 2);
    _Float16* h16    = (_Float16*)alloc((size_t)T_TOT * DIM * 2);
    _Float16* wp16   = (_Float16*)alloc((size_t)DIM * OUTD * 2);
    float*    h2     = (float*)   alloc((size_t)T_TOT * OUTD * 4);
    _Float16* xln16  = (_Float16*)alloc((size_t)T_TOT * OUTD * 2);
    float*    gate   = (float*)   alloc((size_t)T_TOT * NEXP * 4);
    int*      cnt    = (int*)     alloc(NEXP * 4);
    int*      idxl   = (int*)     alloc((size_t)NEXP * T_TOT * 4);
    _Float16* w116   = (_Float16*)alloc((size_t)OUTD * HID * 2);
    _Float16* w216   = (_Float16*)alloc((size_t)HID * OUTD * 2);
    _Float16* ae16   = (_Float16*)alloc((size_t)T_TOT * OUTD * 2);
    _Float16* hid16  = (_Float16*)alloc((size_t)T_TOT * HID * 2);

    // ---- patch embed ----
    k_patchify<<<T_TOT, 224, 0, stream>>>(x, xp16);
    k_pack_wpe<<<KP0, DIM, 0, stream>>>(Wpe, wpe16);
    k_gemm<<<dim3(T_TOT / 64, DIM / 64), 256, 0, stream>>>(
        xp16, KP0, wpe16, DIM, bpe, 0, h, nullptr, DIM, nullptr,
        T_TOT, nullptr, DIM, KP0, nullptr, nullptr, 0, nullptr);
    k_add_pe<<<T_TOT, DIM, 0, stream>>>(h);

    // ---- attention ----
    k_layernorm<<<T_TOT, 256, 0, stream>>>(h, hn16, g1, be1, DIM);
    k_pack_qkv<<<DIM, DIM, 0, stream>>>(Wq, wq16);
    k_pack_qkv<<<DIM, DIM, 0, stream>>>(Wk, wk16);
    k_pack_qkv<<<DIM, DIM, 0, stream>>>(Wv, wv16);
    dim3 gqkv(T_TOT / 64, DIM / 64);
    k_gemm<<<gqkv, 256, 0, stream>>>(hn16, DIM, wq16, DIM, nullptr, 0, qb, nullptr, DIM,
        nullptr, T_TOT, nullptr, DIM, DIM, nullptr, nullptr, 0, nullptr);
    k_gemm<<<gqkv, 256, 0, stream>>>(hn16, DIM, wk16, DIM, nullptr, 0, kb, nullptr, DIM,
        nullptr, T_TOT, nullptr, DIM, DIM, nullptr, nullptr, 0, nullptr);
    k_gemm<<<gqkv, 256, 0, stream>>>(hn16, DIM, wv16, DIM, nullptr, 0, vb, nullptr, DIM,
        nullptr, T_TOT, nullptr, DIM, DIM, nullptr, nullptr, 0, nullptr);
    k_attention<<<Bsz * 8, NTOK, 0, stream>>>(qb, kb, vb, sa16);
    k_f32_to_f16<<<(DIM * DIM / 4) / 256, 256, 0, stream>>>(Wo, wo16, DIM * DIM / 4);
    k_gemm<<<gqkv, 256, 0, stream>>>(sa16, DIM, wo16, DIM, bo, 0, h, nullptr, DIM,
        h /*residual, in-place*/, T_TOT, nullptr, DIM, DIM, nullptr, nullptr, 0, nullptr);

    // ---- projection to OUT ----
    k_f32_to_f16<<<(T_TOT * DIM / 4) / 256, 256, 0, stream>>>(h, h16, T_TOT * DIM / 4);
    k_f32_to_f16<<<(DIM * OUTD / 4) / 256, 256, 0, stream>>>(Wp, wp16, DIM * OUTD / 4);
    k_gemm<<<dim3(T_TOT / 64, OUTD / 64), 256, 0, stream>>>(
        h16, DIM, wp16, OUTD, bp, 0, h2, nullptr, OUTD, nullptr,
        T_TOT, nullptr, OUTD, DIM, nullptr, nullptr, 0, nullptr);

    // ---- MoE layers (routed: only top-2 tokens hit each expert) ----
    auto run_moe = [&](const float* const* P, const float* gam, const float* bet,
                       float* outp, unsigned seed) {
        const float* Wr = P[0]; const float* br = P[1];
        const float* Wn = P[2]; const float* bn = P[3];
        const float* W1 = P[4]; const float* b1 = P[5];
        const float* W2 = P[6]; const float* b2 = P[7];
        k_layernorm<<<T_TOT, 256, 0, stream>>>(h2, xln16, gam, bet, OUTD);
        k_router<<<T_TOT, 256, 0, stream>>>(xln16, Wr, br, Wn, bn, gate, seed);
        (void)hipMemsetAsync(cnt, 0, NEXP * sizeof(int), stream);
        (void)hipMemsetAsync(outp, 0, (size_t)T_TOT * OUTD * sizeof(float), stream);
        k_build_lists<<<T_TOT / 256, 256, 0, stream>>>(gate, cnt, idxl);
        for (int e = 0; e < NEXP; ++e) {
            k_f32_to_f16<<<(OUTD * HID / 4) / 256, 256, 0, stream>>>(
                W1 + (size_t)e * OUTD * HID, w116, OUTD * HID / 4);
            k_f32_to_f16<<<(HID * OUTD / 4) / 256, 256, 0, stream>>>(
                W2 + (size_t)e * HID * OUTD, w216, HID * OUTD / 4);
            k_gather<<<T_TOT, 128, 0, stream>>>(xln16, idxl + e * T_TOT, cnt + e, ae16);
            // hid = relu(Ae @ W1e + b1e)   (f16 out)
            k_gemm<<<dim3(T_TOT / 64, HID / 64), 256, 0, stream>>>(
                ae16, OUTD, w116, HID, b1 + (size_t)e * HID, 1,
                nullptr, hid16, HID, nullptr,
                T_TOT, cnt + e, HID, OUTD, nullptr, nullptr, 0, nullptr);
            // out[tok] += gate * (hid @ W2e + b2e)   (scatter epilogue)
            k_gemm<<<dim3(T_TOT / 64, OUTD / 64), 256, 0, stream>>>(
                hid16, HID, w216, OUTD, b2 + (size_t)e * OUTD, 0,
                nullptr, nullptr, OUTD, nullptr,
                T_TOT, cnt + e, OUTD, HID,
                idxl + e * T_TOT, gate, e, outp);
        }
    };
    run_moe(moe_p[0], g2, be2, out_first, 1001u);
    run_moe(moe_p[1], g3, be3, out_second, 1002u);

    // ---- heads ----
    k_mean<<<dim3(Bsz, OUTD / 256), 256, 0, stream>>>(out_second, out_iv);
    k_cls<<<Bsz, 256, 0, stream>>>(out_iv, Wc, bc, out_cls);
}
